// Conv2DMod_1726576855917
// MI455X (gfx1250) — matrix operand order repaired and checked by reading, not compiled, not run on hardware
//
#include <hip/hip_runtime.h>
#include <hip/hip_bf16.h>

// ---------------------------------------------------------------------------
// Conv2DMod (StyleGAN2 modulated conv) for gfx1250 / MI455X.
//  k0: xpose_kernel    : x f32 [B][C][H][W] -> xT bf16 [B][H][W][C] (LDS transpose)
//  k1: modulate_kernel : modulated+demodulated weights -> bf16 wmod [B][9][O][C]
//  k2: conv_wmma_kernel: implicit GEMM, WG = 128(O) x 128(2 rows of W),
//      K = C*9 reduced with v_wmma_f32_16x16x32_bf16.  A/B tiles staged into
//      double-buffered LDS via global_load_async_to_lds_b128 (ASYNCcnt),
//      software-pipelined: next chunk's DMA overlaps current chunk's WMMAs.
//      +-1 pixel halo B tile shared by all three kw taps.
// ---------------------------------------------------------------------------

typedef __attribute__((ext_vector_type(16))) __bf16 v16bf;
typedef __attribute__((ext_vector_type(8)))  float  v8f;

union FragBF {
    uint4 u[2];   // 2 x 16B ds loads
    v16bf v;      // WMMA operand
};

#define B_    8
#define CIN   512
#define COUT  512
#define HH    64
#define WW    64
#define CK    (CIN * 9)       // 4608
#define LDA   40              // padded LDS stride (shorts): 80B, 16B aligned
#define LDB   40
#define ASZ   (3 * 128 * LDA) // one A buffer (shorts)
#define BSZ   (2 * 66 * LDB)  // one B buffer (shorts)

// round-to-nearest-even f32 -> bf16
__device__ __forceinline__ unsigned short f2bf(float f) {
    unsigned u = __builtin_bit_cast(unsigned, f);
    u += 0x7fffu + ((u >> 16) & 1u);
    return (unsigned short)(u >> 16);
}

// async 16B global -> LDS copy (per-lane), tracked by ASYNCcnt
__device__ __forceinline__ void async_b128(unsigned lds_off, const void* gptr) {
    asm volatile("global_load_async_to_lds_b128 %0, %1, off"
                 :: "v"(lds_off), "v"(gptr) : "memory");
}
__device__ __forceinline__ void wait_async0() {
    asm volatile("s_wait_asynccnt 0x0" ::: "memory");
}

// ---------------------------------------------------------------------------
// k0: transpose+convert x -> xT[b][h][w][c] bf16.  Grid (C/64, H, B), 256 thr.
// ---------------------------------------------------------------------------
__global__ __launch_bounds__(256)
void xpose_kernel(const float* __restrict__ x, unsigned short* __restrict__ xt)
{
    __shared__ unsigned short T[64 * 72];     // 64 c-rows x 64 w, stride 72
    const int tid = threadIdx.x;
    const int c0  = blockIdx.x * 64;
    const int hc  = blockIdx.y;
    const int b   = blockIdx.z;

    {   // load 64x64 f32 tile (w contiguous), convert, store rows to LDS
        const int i  = tid >> 2;              // c row 0..63
        const int wc = (tid & 3) * 16;        // w chunk
        const float* src = x + (((size_t)(b * CIN + c0 + i) * HH) + hc) * WW + wc;
        unsigned short* dst = T + i * 72 + wc;
        unsigned u[8];
        #pragma unroll
        for (int k = 0; k < 8; ++k)
            u[k] = (unsigned)f2bf(src[2*k]) | ((unsigned)f2bf(src[2*k+1]) << 16);
        *(uint4*)(dst)     = make_uint4(u[0], u[1], u[2], u[3]);
        *(uint4*)(dst + 8) = make_uint4(u[4], u[5], u[6], u[7]);
    }
    __syncthreads();
    {   // gather columns (fixed w, c contiguous) and store 32B chunks
        const int w  = tid >> 2;
        const int cc = (tid & 3) * 16;
        unsigned u[8];
        #pragma unroll
        for (int k = 0; k < 8; ++k) {
            unsigned lo = T[(cc + 2*k)     * 72 + w];
            unsigned hi = T[(cc + 2*k + 1) * 72 + w];
            u[k] = lo | (hi << 16);
        }
        unsigned short* dst = xt + (((size_t)(b * HH + hc) * WW) + w) * CIN + c0 + cc;
        *(uint4*)(dst)     = make_uint4(u[0], u[1], u[2], u[3]);
        *(uint4*)(dst + 8) = make_uint4(u[4], u[5], u[6], u[7]);
    }
}

// ---------------------------------------------------------------------------
// k1: one block per (o, b).  wmod[b][r][o][c] = bf16( w[o,c,r]*(y[b,c]+1)*d )
// ---------------------------------------------------------------------------
__global__ __launch_bounds__(256)
void modulate_kernel(const float* __restrict__ weight,   // [O][C][3][3]
                     const float* __restrict__ y,        // [B][C]
                     unsigned short* __restrict__ wmod)  // [B][9][O][C] bf16
{
    const int o   = blockIdx.x;
    const int b   = blockIdx.y;
    const int tid = threadIdx.x;

    __shared__ float red[256];
    __shared__ float dshare;

    const float* wrow = weight + (size_t)o * CK;
    const float* yrow = y + (size_t)b * CIN;

    float s = 0.f;
    for (int e = tid; e < CK; e += 256) {
        int   c = e / 9;
        float f = wrow[e] * (yrow[c] + 1.0f);
        s += f * f;
    }
    red[tid] = s;
    __syncthreads();
    #pragma unroll
    for (int off = 128; off > 0; off >>= 1) {
        if (tid < off) red[tid] += red[tid + off];
        __syncthreads();
    }
    if (tid == 0) dshare = rsqrtf(red[0] + 1e-8f);
    __syncthreads();
    const float d = dshare;

    const size_t plane = (size_t)COUT * CIN;
    const size_t obase = (size_t)b * 9 * plane + (size_t)o * CIN;
    for (int e = tid; e < CK; e += 256) {
        int   r = e >> 9;            // 0..8
        int   c = e & 511;           // 0..511
        float f = wrow[c * 9 + r] * (yrow[c] + 1.0f) * d;
        wmod[obase + (size_t)r * plane + c] = f2bf(f);
    }
}

// ---------------------------------------------------------------------------
// stage one K-chunk (3 kw A-tiles + halo B tile) into LDS buffer via async DMA
// ---------------------------------------------------------------------------
__device__ __forceinline__ void stage_tiles(
    unsigned short* Asb, unsigned short* Bsb,
    const unsigned short* __restrict__ xt,
    const unsigned short* __restrict__ wmod,
    int b, int o0, int h0, int kh, int c0, int tid)
{
    const size_t plane = (size_t)COUT * CIN;
    const size_t abase = (size_t)(b * 9 + kh * 3) * plane + (size_t)o0 * CIN;

    // A: 3 kw-tiles x 128 rows x 32 c = 1536 x 16B async copies
    #pragma unroll
    for (int it = 0; it < 6; ++it) {
        const int ch   = tid + it * 256;
        const int kw   = ch >> 9;            // 0..2
        const int rp   = ch & 511;
        const int row  = rp >> 2;            // 0..127
        const int part = (rp & 3) * 8;       // shorts
        const unsigned short* g =
            wmod + abase + (size_t)kw * plane + (size_t)row * CIN + c0 + part;
        async_b128((unsigned)(size_t)(Asb + (kw * 128 + row) * LDA + part), g);
    }

    // B: 2 rows x 66 halo pixels x 32 c = 528 x 16B
    for (int ch = tid; ch < 528; ch += 256) {
        const int r2   = (ch >= 264) ? 1 : 0;
        const int rem  = ch - r2 * 264;
        const int p    = rem >> 2;           // halo pixel 0..65
        const int part = (rem & 3) * 8;
        const int hh   = h0 + r2 + kh - 1;
        const int w    = p - 1;
        unsigned short* l = Bsb + (r2 * 66 + p) * LDB + part;
        if ((unsigned)hh < (unsigned)HH && (unsigned)w < (unsigned)WW) {
            const unsigned short* g =
                xt + (((size_t)(b * HH + hh) * WW) + w) * CIN + c0 + part;
            async_b128((unsigned)(size_t)l, g);
        } else {
            *(uint4*)l = make_uint4(0u, 0u, 0u, 0u);  // zero padding
        }
    }
}

// ---------------------------------------------------------------------------
// k2: implicit-GEMM conv.  grid = (H/2=32, O/128=4, B=8), 256 threads (8 waves).
// Wave grid 4(M) x 2(N); wave tile 32 x 64 -> acc[2][4], 24 WMMA per K-chunk.
// Double-buffered LDS, async DMA pipelined against WMMA compute.
// ---------------------------------------------------------------------------
__global__ __launch_bounds__(256)
void conv_wmma_kernel(const unsigned short* __restrict__ xt,   // [B][H][W][C] bf16
                      const unsigned short* __restrict__ wmod, // [B][9][O][C] bf16
                      float* __restrict__ out)                 // [B][O][H][W] f32
{
    __shared__ unsigned short As[2][ASZ];   // per-kw A tiles: 128 o x 32 c, x2 buf
    __shared__ unsigned short Bs[2][BSZ];   // 2 rows x 66 halo x 32 c, x2 buf

    const int h0   = blockIdx.x * 2;       // first of 2 output rows
    const int o0   = blockIdx.y * 128;     // O tile base
    const int b    = blockIdx.z;
    const int tid  = threadIdx.x;
    const int lane = tid & 31;
    const int wave = tid >> 5;
    const int mW   = (wave & 3) * 32;      // wave M offset in 128
    const int nW   = (wave >> 2) * 64;     // wave N offset in 128
    const int l15  = lane & 15;
    const int lhi  = lane >> 4;            // half-wave

    v8f acc[2][4];
    #pragma unroll
    for (int i = 0; i < 2; ++i)
        #pragma unroll
        for (int j = 0; j < 4; ++j)
            acc[i][j] = {};

    // prologue: stage chunk 0 into buffer 0
    stage_tiles(As[0], Bs[0], xt, wmod, b, o0, h0, /*kh=*/0, /*c0=*/0, tid);

    int buf = 0;
    for (int itr = 0; itr < 48; ++itr) {         // 3 kh x 16 c-chunks
        wait_async0();        // my async writes into As[buf]/Bs[buf] landed
        __syncthreads();      // all waves: buf published, buf^1 free for reuse

        if (itr + 1 < 48) {   // pipeline: DMA next chunk while computing this one
            const int kh1 = (itr + 1) >> 4;
            const int c01 = ((itr + 1) & 15) * 32;
            stage_tiles(As[buf ^ 1], Bs[buf ^ 1], xt, wmod, b, o0, h0, kh1, c01, tid);
        }

        // ---- compute: 3 kw taps share the halo B tile (shifted columns)
        const unsigned short* Asb = As[buf];
        const unsigned short* Bsb = Bs[buf];
        #pragma unroll
        for (int kw = 0; kw < 3; ++kw) {
            FragBF fa[2], fb[4];
            #pragma unroll
            for (int i = 0; i < 2; ++i) {
                const unsigned short* p =
                    Asb + (kw * 128 + mW + 16 * i + l15) * LDA + lhi * 8;
                fa[i].u[0] = *(const uint4*)(p);
                fa[i].u[1] = *(const uint4*)(p + 16);
            }
            #pragma unroll
            for (int j = 0; j < 4; ++j) {
                const int n  = nW + 16 * j + l15;   // 0..127
                const int r2 = n >> 6;
                const int pp = (n & 63) + kw;       // shifted halo pixel
                const unsigned short* p =
                    Bsb + (r2 * 66 + pp) * LDB + lhi * 8;
                fb[j].u[0] = *(const uint4*)(p);
                fb[j].u[1] = *(const uint4*)(p + 16);
            }
            #pragma unroll
            for (int i = 0; i < 2; ++i)
                #pragma unroll
                for (int j = 0; j < 4; ++j)
                    acc[i][j] = __builtin_amdgcn_wmma_f32_16x16x32_bf16(
                        false, fa[i].v, false, fb[j].v,
                        (short)0, acc[i][j], false, false);
        }
        buf ^= 1;
    }

    // ---- epilogue: C/D layout: VGPR rr -> row rr + 8*lhi, lane l15 -> col.
    // Output is write-once: non-temporal stores keep L2 for xT/wmod tiles.
    const size_t ob = (size_t)b * COUT * (HH * WW);
    #pragma unroll
    for (int i = 0; i < 2; ++i)
        #pragma unroll
        for (int j = 0; j < 4; ++j) {
            const int n    = nW + 16 * j + l15;
            const int hrow = n >> 6;
            const int w    = n & 63;
            #pragma unroll
            for (int rr = 0; rr < 8; ++rr) {
                const int mm = o0 + mW + 16 * i + rr + 8 * lhi;
                __builtin_nontemporal_store(
                    acc[i][j][rr],
                    &out[ob + (size_t)mm * (HH * WW) + (h0 + hrow) * WW + w]);
            }
        }
}

// ---------------------------------------------------------------------------
extern "C" void kernel_launch(void* const* d_in, const int* in_sizes, int n_in,
                              void* d_out, int out_size, void* d_ws, size_t ws_size,
                              hipStream_t stream) {
    const float* x      = (const float*)d_in[0];   // [8,512,64,64]
    const float* y      = (const float*)d_in[1];   // [8,512]
    const float* weight = (const float*)d_in[2];   // [512,512,3,3]
    float* out          = (float*)d_out;           // [8,512,64,64]

    unsigned short* wmod = (unsigned short*)d_ws;                    // 36 MB bf16
    unsigned short* xT   = wmod + (size_t)B_ * 9 * COUT * CIN;       // 32 MB bf16

    dim3 g0(CIN / 64, HH, B_);
    xpose_kernel<<<g0, 256, 0, stream>>>(x, xT);

    dim3 g1(COUT, B_);
    modulate_kernel<<<g1, 256, 0, stream>>>(weight, y, wmod);

    dim3 g2(HH / 2, COUT / 128, B_);
    conv_wmma_kernel<<<g2, 256, 0, stream>>>(xT, wmod, out);
}
